// CausalSelfAttention_60284160967096
// MI455X (gfx1250) — matrix operand order repaired and checked by reading, not compiled, not run on hardware
//
#include <hip/hip_runtime.h>
#include <hip/hip_bf16.h>
#include <math.h>

#define DIMN 2048
#define NHE  16
#define NKVH 4
#define HDIM 128
#define GRP  (NHE / NKVH)
#define BB   2
#define TT   2048
#define BTT  (BB * TT)
#define KVD  (NKVH * HDIM) /* 512 */
#define RMS_EPS_F 1.1920928955078125e-07f

typedef __attribute__((ext_vector_type(16))) __bf16 v16bf;
typedef __attribute__((ext_vector_type(8)))  float  v8f;
typedef __attribute__((ext_vector_type(4)))  unsigned int v4u;
typedef __attribute__((ext_vector_type(8)))  int    v8i;
typedef __attribute__((ext_vector_type(4)))  int    v4i;

union Frag { v16bf v; uint4 u[2]; };

__device__ __forceinline__ v8f wmma_bf16(v16bf a, v16bf b, v8f c) {
  return __builtin_amdgcn_wmma_f32_16x16x32_bf16(false, a, false, b, (short)0, c,
                                                 false, false);
}

// Issue a TDM load of a 2-D bf16 tile (tile_w x tile_h, row stride K elems)
// from global `gsrc` into LDS byte-offset `lds_addr`. One issue per wave.
__device__ __forceinline__ void tdm_load_tile_bf16(const __bf16* gsrc,
                                                   unsigned lds_addr,
                                                   unsigned tile_w,
                                                   unsigned tile_h,
                                                   unsigned K) {
  unsigned long long ga = (unsigned long long)(const void*)gsrc;
  v4u g0;
  g0[0] = 1u;                    // count=1, is_restore=0, gather off
  g0[1] = lds_addr;              // lds_addr [63:32]
  g0[2] = (unsigned)ga;          // global_addr [95:64]
  g0[3] = (unsigned)((ga >> 32) & 0x01FFFFFFull) | 0x80000000u;  // addr hi | type=2
  v8i g1;
  g1[0] = 0x00010000;                            // wg_mask=0, data_size=1 (2B)
  g1[1] = (int)((K & 0xFFFFu) << 16);            // tensor_dim0 lo16
  g1[2] = (int)((K >> 16) & 0xFFFFu);            // tensor_dim0 hi16 | dim1 lo16(=0)
  g1[3] = (int)(16u | (tile_w << 16));           // tensor_dim1 hi16 (=16<<16 rows) | tile_dim0
  g1[4] = (int)tile_h;                           // tile_dim1 | tile_dim2=0
  g1[5] = (int)K;                                // tensor_dim0_stride lo32
  g1[6] = 0;                                     // stride0 hi16 | stride1 lo16
  g1[7] = 0;                                     // stride1 hi32
  v4i z4; z4[0] = 0; z4[1] = 0; z4[2] = 0; z4[3] = 0;
  v8i z8;
#pragma unroll
  for (int i = 0; i < 8; ++i) z8[i] = 0;
  // 6-arg form (clang-23 / therock headers): groups 0..3 + extra group + cpol
  __builtin_amdgcn_tensor_load_to_lds(g0, g1, z4, z4, z8, 0);
}

// ---------------------------------------------------------------- utilities
__global__ void f32_to_bf16_kernel(const float* __restrict__ in,
                                   __bf16* __restrict__ out, int n) {
  int i = blockIdx.x * blockDim.x + threadIdx.x;
  if (i < n) out[i] = (__bf16)in[i];
}

// BitLinear weight preparation: one block per output row.
__global__ __launch_bounds__(256) void quant_rows_kernel(
    const float* __restrict__ W, __bf16* __restrict__ Wb,
    const float* __restrict__ sfp, int K) {
  const int n = blockIdx.x, tid = threadIdx.x;
  const float* wr = W + (size_t)n * K;
  float s = 0.f;
  for (int i = tid; i < K; i += 256) s += fabsf(wr[i]);
  __shared__ float red[256];
  red[tid] = s;
  __syncthreads();
  for (int st = 128; st > 0; st >>= 1) {
    if (tid < st) red[tid] += red[tid + st];
    __syncthreads();
  }
  const float scale = fmaxf(red[0] / (float)K, 1e-8f);
  const float sf = sfp[0];
  for (int i = tid; i < K; i += 256) {
    float w = wr[i];
    float wq = (fabsf(w) > 0.7f * scale) ? copysignf(scale, w) : 0.0f;
    Wb[(size_t)n * K + i] = (__bf16)(w + sf * (wq - w));
  }
}

// ---------------------------------------------------------------- WMMA GEMM
// O[M,N] = X[M,K](bf16) @ W[N,K]^T (bf16), f32 out.
// Block = 8 waves = 128x64 tile. Weight 64x32 k-slab staged in LDS by the
// Tensor Data Mover (double-buffered, issued by wave 0, TENSORcnt-synced);
// all 8 waves consume B fragments from LDS (8x L2 traffic reduction).
__global__ __launch_bounds__(256) void gemm_xwT_kernel(
    const __bf16* __restrict__ X, const __bf16* __restrict__ W,
    float* __restrict__ O, int M, int N, int K) {
  const int lane = threadIdx.x & 31;
  const int wv   = threadIdx.x >> 5;
  const int m0 = blockIdx.y * 128 + wv * 16;
  const int n0 = blockIdx.x * 64;
  const int ml = lane & 15;
  const int kh = lane >> 4;

  // double-buffered weight tile: 64 rows x 32 bf16 (row stride 64B)
  __shared__ __align__(128) __bf16 wtile[2][64 * 32];
  const unsigned lds0 =
      (unsigned)(unsigned long long)(const void*)&wtile[0][0];
  const unsigned lds1 =
      (unsigned)(unsigned long long)(const void*)&wtile[1][0];

  const int ksteps = K / 32;
  if (wv == 0) {
    tdm_load_tile_bf16(W + (size_t)n0 * K, lds0, 32u, 64u, (unsigned)K);
    __builtin_amdgcn_s_wait_tensorcnt(0);
  }
  __syncthreads();

  v8f acc[4] = {};
  const __bf16* xrow = X + (size_t)(m0 + ml) * K;
  for (int ks = 0; ks < ksteps; ++ks) {
    const int k0 = ks * 32;
    const int cur = ks & 1;
    if (wv == 0 && ks + 1 < ksteps)  // prefetch next slab via TDM
      tdm_load_tile_bf16(W + (size_t)n0 * K + (k0 + 32), cur ? lds0 : lds1,
                         32u, 64u, (unsigned)K);

    Frag a;  // 16-bit A 16x32 layout: lane holds K[kh*8..+7] and K[16+kh*8..+7]
    const __bf16* xp = xrow + k0 + kh * 8;
    a.u[0] = *(const uint4*)xp;
    a.u[1] = *(const uint4*)(xp + 16);
    __builtin_prefetch(xp + 128, 0, 1);  // global_prefetch: next A cacheline

    const __bf16* wt = &wtile[cur][0];
#pragma unroll
    for (int nt = 0; nt < 4; ++nt) {
      Frag bf;  // B 32x16 layout: lane (col n) holds K[kh*8..+7] and K[16+kh*8..+7]
      const __bf16* wp = wt + (size_t)(nt * 16 + ml) * 32 + kh * 8;
      bf.u[0] = *(const uint4*)wp;
      bf.u[1] = *(const uint4*)(wp + 16);
      acc[nt] = wmma_bf16(a.v, bf.v, acc[nt]);
    }

    if (wv == 0 && ks + 1 < ksteps)
      __builtin_amdgcn_s_wait_tensorcnt(0);  // next buffer landed
    __syncthreads();                         // all waves done with cur buffer
  }

  const int rbase = m0 + 8 * kh;  // C layout: VGPR r -> row r + 8*(lane>=16)
#pragma unroll
  for (int nt = 0; nt < 4; ++nt)
#pragma unroll
    for (int r = 0; r < 8; ++r)
      O[(size_t)(rbase + r) * N + n0 + nt * 16 + ml] = acc[nt][r];
}

// ---------------------------------------------------- RMS-norm + RoPE (+gain)
__global__ __launch_bounds__(HDIM) void rmsnorm_rope_kernel(
    const float* __restrict__ G, __bf16* __restrict__ O,
    const float* __restrict__ gain, int nheads) {
  const int idx = blockIdx.x;
  const int h = idx % nheads;
  const int t = (idx / nheads) % TT;
  const int b = idx / (nheads * TT);
  const int d = threadIdx.x;
  float val = G[(size_t)(b * TT + t) * (nheads * HDIM) + h * HDIM + d];
  __shared__ float red[HDIM];
  __shared__ float sv[HDIM];
  red[d] = val * val;
  __syncthreads();
  for (int st = 64; st > 0; st >>= 1) {
    if (d < st) red[d] += red[d + st];
    __syncthreads();
  }
  const float rinv = rsqrtf(red[0] * (1.0f / HDIM) + RMS_EPS_F);
  sv[d] = val * rinv;
  __syncthreads();
  if (d < HDIM / 2) {
    const float invf = __expf(-(float)d * 0.14391156831f);  // ln(1e4)/64
    const float ang = (float)t * invf;
    float s, c;
    __sincosf(ang, &s, &c);
    const float x1 = sv[d], x2 = sv[d + HDIM / 2];
    const float g = gain ? gain[h] : 1.0f;
    const size_t base = ((size_t)(b * nheads + h) * TT + t) * HDIM;
    O[base + d]            = (__bf16)((x1 * c + x2 * s) * g);
    O[base + d + HDIM / 2] = (__bf16)((x2 * c - x1 * s) * g);
  }
}

// -------------------------------------------------------------- V preparation
__global__ __launch_bounds__(HDIM) void v_prep_kernel(
    const float* __restrict__ G, __bf16* __restrict__ VT,
    float* __restrict__ VN) {
  const int idx = blockIdx.x;
  const int h = idx % NKVH;
  const int t = (idx / NKVH) % TT;
  const int b = idx / (NKVH * TT);
  const int d = threadIdx.x;
  float val = G[(size_t)(b * TT + t) * KVD + h * HDIM + d];
  __shared__ float red[HDIM];
  red[d] = val * val;
  __syncthreads();
  for (int st = 64; st > 0; st >>= 1) {
    if (d < st) red[d] += red[d + st];
    __syncthreads();
  }
  const float inv = 1.0f / fmaxf(sqrtf(red[0]), 1e-12f);
  VN[((size_t)(b * NKVH + h) * TT + t) * HDIM + d] = val * inv;
  VT[((size_t)(b * NKVH + h) * HDIM + d) * TT + t] = (__bf16)val;
}

// ------------------------------------------------------------ flash attention
__global__ __launch_bounds__(256) void attn_flash_kernel(
    const __bf16* __restrict__ Q, const __bf16* __restrict__ Kb,
    const __bf16* __restrict__ VT, float* __restrict__ Y) {
  const int lane = threadIdx.x & 31;
  const int wv   = threadIdx.x >> 5;
  const int bh = blockIdx.y;
  const int b = bh / NHE, h = bh % NHE, kv = h / GRP;
  const int q0 = blockIdx.x * 128 + wv * 16;
  const int ml = lane & 15, kh = lane >> 4;
  const __bf16* Qp = Q  + ((size_t)bh * TT + q0) * HDIM;
  const __bf16* Kp = Kb + ((size_t)(b * NKVH + kv) * TT) * HDIM;
  const __bf16* Vp = VT + ((size_t)(b * NKVH + kv) * HDIM) * TT;

  Frag qf[4];
#pragma unroll
  for (int f = 0; f < 4; ++f) {
    const __bf16* p = Qp + (size_t)ml * HDIM + f * 32 + kh * 8;
    qf[f].u[0] = *(const uint4*)p;
    qf[f].u[1] = *(const uint4*)(p + 16);
  }
  v8f o[8] = {};
  float rm[8], rl[8];
#pragma unroll
  for (int r = 0; r < 8; ++r) { rm[r] = -3.0e38f; rl[r] = 0.f; }

  __shared__ __align__(16) __bf16 pst[8][16 * 32];
  __bf16* ps = pst[wv];
  const float scale = 0.08838834764831845f;  // 1/sqrt(128)
  const int qhi = q0 + 15;

  for (int c = 0; c <= qhi; c += 32) {
    v8f s0 = {}, s1 = {};
#pragma unroll
    for (int f = 0; f < 4; ++f) {
      Frag k0f, k1f;
      const __bf16* kp0 = Kp + (size_t)(c + ml) * HDIM + f * 32 + kh * 8;
      k0f.u[0] = *(const uint4*)kp0;
      k0f.u[1] = *(const uint4*)(kp0 + 16);
      const __bf16* kp1 = kp0 + (size_t)16 * HDIM;
      k1f.u[0] = *(const uint4*)kp1;
      k1f.u[1] = *(const uint4*)(kp1 + 16);
      s0 = wmma_bf16(qf[f].v, k0f.v, s0);
      s1 = wmma_bf16(qf[f].v, k1f.v, s1);
    }
    float corr[8];
#pragma unroll
    for (int r = 0; r < 8; ++r) {
      const int qrow = q0 + 8 * kh + r;
      float a0 = (c + ml      <= qrow) ? s0[r] * scale : -3.0e38f;
      float a1 = (c + 16 + ml <= qrow) ? s1[r] * scale : -3.0e38f;
      float mx = fmaxf(a0, a1);
#pragma unroll
      for (int off = 1; off < 16; off <<= 1)
        mx = fmaxf(mx, __shfl_xor(mx, off, 32));
      const float mnew = fmaxf(rm[r], mx);
      corr[r] = __expf(rm[r] - mnew);
      const float e0 = __expf(a0 - mnew);
      const float e1 = __expf(a1 - mnew);
      float sm = e0 + e1;
#pragma unroll
      for (int off = 1; off < 16; off <<= 1) sm += __shfl_xor(sm, off, 32);
      rl[r] = rl[r] * corr[r] + sm;
      rm[r] = mnew;
      ps[(8 * kh + r) * 32 + ml]      = (__bf16)e0;
      ps[(8 * kh + r) * 32 + 16 + ml] = (__bf16)e1;
    }
    Frag pf;
    pf.u[0] = *(const uint4*)&ps[ml * 32 + kh * 8];
    pf.u[1] = *(const uint4*)&ps[ml * 32 + 16 + kh * 8];
#pragma unroll
    for (int j = 0; j < 8; ++j) {
#pragma unroll
      for (int r = 0; r < 8; ++r) o[j][r] *= corr[r];
      Frag vb;
      const __bf16* vp = Vp + (size_t)(j * 16 + ml) * TT + c + kh * 8;
      vb.u[0] = *(const uint4*)vp;
      vb.u[1] = *(const uint4*)(vp + 16);
      o[j] = wmma_bf16(pf.v, vb.v, o[j]);
    }
  }
#pragma unroll
  for (int j = 0; j < 8; ++j)
#pragma unroll
    for (int r = 0; r < 8; ++r)
      Y[((size_t)bh * TT + q0 + 8 * kh + r) * HDIM + j * 16 + ml] =
          o[j][r] / rl[r];
}

// -------------------------------------------------- v-projection removal
__global__ __launch_bounds__(HDIM) void vproj_remove_kernel(
    const float* __restrict__ Y, const float* __restrict__ VN,
    __bf16* __restrict__ Yb) {
  const int idx = blockIdx.x;
  const int h = idx % NHE;
  const int t = (idx / NHE) % TT;
  const int b = idx / (NHE * TT);
  const int kv = h / GRP;
  const int d = threadIdx.x;
  const float y = Y[((size_t)(b * NHE + h) * TT + t) * HDIM + d];
  const float w = VN[((size_t)(b * NKVH + kv) * TT + t) * HDIM + d];
  __shared__ float red[HDIM];
  red[d] = y * w;
  __syncthreads();
  for (int st = 64; st > 0; st >>= 1) {
    if (d < st) red[d] += red[d + st];
    __syncthreads();
  }
  const float dot = red[0];
  Yb[(size_t)(b * TT + t) * DIMN + h * HDIM + d] = (__bf16)(y - dot * w);
}

// ------------------------------------------------------------------- launch
extern "C" void kernel_launch(void* const* d_in, const int* in_sizes, int n_in,
                              void* d_out, int out_size, void* d_ws,
                              size_t ws_size, hipStream_t stream) {
  (void)in_sizes; (void)n_in; (void)out_size; (void)ws_size;
  const float* x    = (const float*)d_in[0];
  const float* sfp  = (const float*)d_in[1];
  const float* w_q  = (const float*)d_in[2];
  const float* w_k  = (const float*)d_in[3];
  const float* w_v  = (const float*)d_in[4];
  const float* w_p  = (const float*)d_in[5];
  const float* gain = (const float*)d_in[6];

  char* ws = (char*)d_ws;
  size_t off = 0;
  auto take = [&](size_t bytes) -> char* {
    char* p = ws + off;
    off = (off + bytes + 255) & ~(size_t)255;
    return p;
  };
  __bf16* x_bf  = (__bf16*)take((size_t)BTT * DIMN * 2);
  __bf16* wq_bf = (__bf16*)take((size_t)DIMN * DIMN * 2);
  __bf16* wk_bf = (__bf16*)take((size_t)KVD * DIMN * 2);
  __bf16* wv_bf = (__bf16*)take((size_t)KVD * DIMN * 2);
  __bf16* wp_bf = (__bf16*)take((size_t)DIMN * DIMN * 2);
  float*  gq    = (float*)take((size_t)BTT * DIMN * 4);
  float*  gk    = (float*)take((size_t)BTT * KVD * 4);
  float*  gv    = (float*)take((size_t)BTT * KVD * 4);
  __bf16* q_bf  = (__bf16*)take((size_t)BB * NHE * TT * HDIM * 2);
  __bf16* k_bf  = (__bf16*)take((size_t)BB * NKVH * TT * HDIM * 2);
  __bf16* vT_bf = (__bf16*)take((size_t)BB * NKVH * HDIM * TT * 2);
  float*  vn    = (float*)take((size_t)BB * NKVH * TT * HDIM * 4);
  float*  y_f   = gq;    // alias: gq fully consumed before attention writes
  __bf16* y_bf  = x_bf;  // alias: x_bf fully consumed before proj-removal

  f32_to_bf16_kernel<<<(BTT * DIMN) / 256, 256, 0, stream>>>(x, x_bf,
                                                             BTT * DIMN);
  quant_rows_kernel<<<DIMN, 256, 0, stream>>>(w_q, wq_bf, sfp, DIMN);
  quant_rows_kernel<<<KVD,  256, 0, stream>>>(w_k, wk_bf, sfp, DIMN);
  quant_rows_kernel<<<KVD,  256, 0, stream>>>(w_v, wv_bf, sfp, DIMN);
  quant_rows_kernel<<<DIMN, 256, 0, stream>>>(w_p, wp_bf, sfp, DIMN);

  gemm_xwT_kernel<<<dim3(DIMN / 64, BTT / 128), 256, 0, stream>>>(
      x_bf, wq_bf, gq, BTT, DIMN, DIMN);
  gemm_xwT_kernel<<<dim3(KVD / 64, BTT / 128), 256, 0, stream>>>(
      x_bf, wk_bf, gk, BTT, KVD, DIMN);
  gemm_xwT_kernel<<<dim3(KVD / 64, BTT / 128), 256, 0, stream>>>(
      x_bf, wv_bf, gv, BTT, KVD, DIMN);

  rmsnorm_rope_kernel<<<BB * TT * NHE,  HDIM, 0, stream>>>(gq, q_bf, gain, NHE);
  rmsnorm_rope_kernel<<<BB * TT * NKVH, HDIM, 0, stream>>>(gk, k_bf, nullptr,
                                                           NKVH);
  v_prep_kernel<<<BB * TT * NKVH, HDIM, 0, stream>>>(gv, vT_bf, vn);

  attn_flash_kernel<<<dim3(TT / 128, BB * NHE), 256, 0, stream>>>(
      q_bf, k_bf, vT_bf, y_f);

  vproj_remove_kernel<<<BB * TT * NHE, HDIM, 0, stream>>>(y_f, vn, y_bf);

  gemm_xwT_kernel<<<dim3(DIMN / 64, BTT / 128), 256, 0, stream>>>(
      y_bf, wp_bf, (float*)d_out, BTT, DIMN, DIMN);
}
